// MambaDecoder_32040456028654
// MI455X (gfx1250) — compile-verified, matrix operand order checked by reading, not run on hardware
//
#include <hip/hip_runtime.h>
#include <hip/hip_bf16.h>
#include <math.h>

// ---- model constants (from reference) ----
#define BATCH   2
#define TLEN    512
#define ND      256
#define STRIDE  4
#define KLEN    14
#define DM      3584          // ND*KLEN
#define DI      7168          // 2*DM
#define XZ_N    (2*DI)        // 14336
#define DSTATE  16
#define DCONV   4
#define DTR     224
#define NCLS    41
#define LOUT    125
#define ROWS    250           // BATCH*LOUT
#define ROWS_PAD 256

typedef __attribute__((ext_vector_type(16))) _Float16 v16h;
typedef __attribute__((ext_vector_type(8)))  _Float16 v8h;
typedef __attribute__((ext_vector_type(8)))  float    v8f;

// ---------------------------------------------------------------------------
// 16-bit A/B WMMA fragment layout (wave32): lane&15 = matrix row (A) / col (B);
// lane>>4 = K-half group g. Per-lane K indices: {g*8..g*8+7, 16+g*8..16+g*8+7}
// = two contiguous runs.
// ---------------------------------------------------------------------------

// fp32 source (weight stream): four b128 loads + packed cvt to f16.
__device__ __forceinline__ v16h load_frag_f32(const float* __restrict__ p, int g) {
  float4 a = *reinterpret_cast<const float4*>(p + g * 8);
  float4 b = *reinterpret_cast<const float4*>(p + g * 8 + 4);
  float4 c = *reinterpret_cast<const float4*>(p + 16 + g * 8);
  float4 d = *reinterpret_cast<const float4*>(p + 16 + g * 8 + 4);
  v16h f;
  f[0]  = (_Float16)a.x; f[1]  = (_Float16)a.y; f[2]  = (_Float16)a.z; f[3]  = (_Float16)a.w;
  f[4]  = (_Float16)b.x; f[5]  = (_Float16)b.y; f[6]  = (_Float16)b.z; f[7]  = (_Float16)b.w;
  f[8]  = (_Float16)c.x; f[9]  = (_Float16)c.y; f[10] = (_Float16)c.z; f[11] = (_Float16)c.w;
  f[12] = (_Float16)d.x; f[13] = (_Float16)d.y; f[14] = (_Float16)d.z; f[15] = (_Float16)d.w;
  return f;
}

// f16 source (pre-converted activations): two b128 loads, no VALU conversion.
__device__ __forceinline__ v16h load_frag_f16(const _Float16* __restrict__ p, int g) {
  v8h lo = *reinterpret_cast<const v8h*>(p + g * 8);
  v8h hi = *reinterpret_cast<const v8h*>(p + 16 + g * 8);
  v16h f;
#pragma unroll
  for (int e = 0; e < 8; ++e) { f[e] = lo[e]; f[8 + e] = hi[e]; }
  return f;
}

// ---------------------------------------------------------------------------
// C = A(256 x K f16, row-major, lda) @ W(N x K fp32, row-major)^T.
// Block = 4 waves; each wave owns one 16-wide N tile and register-blocks
// 8 M-tiles (128 rows); grid.y = 2 covers the 256 padded rows. One converted
// weight fragment feeds 8 v_wmma_f32_16x16x32_f16.
// epi==1: C = softplus(C + bias[n]); Ch (optional) = f16 mirror of C.
// ---------------------------------------------------------------------------
__global__ void __launch_bounds__(128)
gemm_f16_wmma(const _Float16* __restrict__ A, const float* __restrict__ W,
              float* __restrict__ C, _Float16* __restrict__ Ch,
              const float* __restrict__ bias,
              int K, int N, int lda, int ldw, int epi) {
  const int wave = threadIdx.x >> 5;
  const int lane = threadIdx.x & 31;
  const int rc   = lane & 15;           // row (A) / col (B,C) within tile
  const int g    = lane >> 4;           // K-half group
  const int nBase = (blockIdx.x * 4 + wave) * 16;
  const int mBase = blockIdx.y * 128;

  const float*    wrow  = W + (size_t)(nBase + rc) * ldw;
  const _Float16* arow0 = A + (size_t)(mBase + rc) * lda;

  v8f acc[8] = {};
  for (int kk = 0; kk < K; kk += 32) {
    __builtin_prefetch(wrow + kk + 32, 0, 1);        // global_prefetch on weight stream
    v16h bfrag = load_frag_f32(wrow + kk, g);
    const _Float16* ap = arow0 + kk;
#pragma unroll
    for (int mt = 0; mt < 8; ++mt) {
      v16h afrag = load_frag_f16(ap + (size_t)mt * 16 * lda, g);
      acc[mt] = __builtin_amdgcn_wmma_f32_16x16x32_f16(
          false, afrag, false, bfrag, (short)0, acc[mt], false, false);
    }
  }

  // f32 C/D layout: reg r -> row r + g*8 ; col = lane&15
  const int cc = nBase + rc;
#pragma unroll
  for (int mt = 0; mt < 8; ++mt) {
#pragma unroll
    for (int r = 0; r < 8; ++r) {
      int rr = mBase + mt * 16 + g * 8 + r;
      float v = acc[mt][r];
      if (epi == 1) {
        v += bias[cc];
        v = (v > 20.f) ? v : log1pf(__expf(v));      // softplus
      }
      C[(size_t)rr * N + cc] = v;
      if (Ch) Ch[(size_t)rr * N + cc] = (_Float16)v;
    }
  }
}

// ---- day-specific linear + softsign:  t[b,ti,k] = softsign(in . W[day] + b) ----
__global__ void __launch_bounds__(256)
day_transform(const float* __restrict__ in, const float* __restrict__ day_w,
              const float* __restrict__ day_b, const int* __restrict__ dayIdx,
              float* __restrict__ t) {
  int b = blockIdx.y, ti = blockIdx.x, k = threadIdx.x;
  __shared__ float srow[ND];
  srow[k] = in[((size_t)b * TLEN + ti) * ND + k];
  __syncthreads();
  int day = dayIdx[b];
  const float* W = day_w + (size_t)day * ND * ND;
  float acc = day_b[(size_t)day * ND + k];
  for (int d = 0; d < ND; ++d) acc += srow[d] * W[d * ND + k];
  acc = acc / (1.f + fabsf(acc));
  t[((size_t)b * TLEN + ti) * ND + k] = acc;
}

// ---- unfold: h[b*125+l, nd*14+k] = t[b, l*4+k, nd]; pad rows -> 0 ----
__global__ void __launch_bounds__(256)
unfold_kernel(const float* __restrict__ t, float* __restrict__ h) {
  int r = blockIdx.x;
  for (int c = threadIdx.x; c < DM; c += 256) {
    float v = 0.f;
    if (r < ROWS) {
      int b = r / LOUT, l = r % LOUT;
      int nd = c / KLEN, k = c % KLEN;
      v = t[((size_t)b * TLEN + (l * STRIDE + k)) * ND + nd];
    }
    h[(size_t)r * DM + c] = v;
  }
}

// ---- residual-add + layernorm; writes fp32 hn (for head) + f16 hn_h (GEMM A) ----
__global__ void __launch_bounds__(256)
ln_residual(const float* __restrict__ h, float* __restrict__ residual,
            float* __restrict__ hn, _Float16* __restrict__ hn_h,
            const float* __restrict__ w, const float* __restrict__ b, int first) {
  __shared__ float s1[256], s2[256];
  int r = blockIdx.x, tid = threadIdx.x;
  if (r >= ROWS) {                       // pad rows -> exact zeros
    for (int k = tid; k < DM; k += 256) {
      size_t idx = (size_t)r * DM + k;
      residual[idx] = 0.f;
      hn[idx] = 0.f;
      hn_h[idx] = (_Float16)0.f;
    }
    return;
  }
  float sum = 0.f, sq = 0.f;
  for (int k = tid; k < DM; k += 256) {
    size_t idx = (size_t)r * DM + k;
    float v = h[idx];
    if (!first) v += residual[idx];
    residual[idx] = v;
    sum += v; sq += v * v;
  }
  s1[tid] = sum; s2[tid] = sq;
  __syncthreads();
  for (int s = 128; s > 0; s >>= 1) {
    if (tid < s) { s1[tid] += s1[tid + s]; s2[tid] += s2[tid + s]; }
    __syncthreads();
  }
  float mu  = s1[0] / DM;
  float var = s2[0] / DM - mu * mu;
  float rs  = rsqrtf(var + 1e-5f);
  for (int k = tid; k < DM; k += 256) {
    size_t idx = (size_t)r * DM + k;
    float v = (residual[idx] - mu) * rs * w[k] + b[k];
    hn[idx] = v;
    hn_h[idx] = (_Float16)v;
  }
}

// ---- depthwise causal conv (DCONV=4) + SiLU; fp32 (scan input) + f16 (GEMM A) ----
__global__ void __launch_bounds__(256)
conv_silu(const float* __restrict__ xz, const float* __restrict__ cw,
          const float* __restrict__ cb, float* __restrict__ xbuf,
          _Float16* __restrict__ xbuf_h) {
  int r = blockIdx.x;
  int valid = r < ROWS;
  int b = r / LOUT, l = r % LOUT;
  for (int d = threadIdx.x; d < DI; d += 256) {
    float v = 0.f;
    if (valid) {
      v = cb[d];
#pragma unroll
      for (int j = 0; j < DCONV; ++j) {
        int ll = l + j - (DCONV - 1);
        if (ll >= 0) v += xz[(size_t)(b * LOUT + ll) * XZ_N + d] * cw[d * DCONV + j];
      }
      v = v / (1.f + __expf(-v));                // silu
    }
    xbuf[(size_t)r * DI + d] = v;
    xbuf_h[(size_t)r * DI + d] = (_Float16)v;
  }
}

// ---- selective scan + skip (D) + z-gating; emits f16 (only consumer = GEMM) ----
__global__ void __launch_bounds__(256)
scan_gate(const float* __restrict__ xdbl, const float* __restrict__ delta,
          const float* __restrict__ xbuf, const float* __restrict__ xz,
          const float* __restrict__ Alog, const float* __restrict__ Dp,
          _Float16* __restrict__ ybuf_h) {
  __shared__ float sB[LOUT * DSTATE];
  __shared__ float sC[LOUT * DSTATE];
  int b = blockIdx.y;
  int d = blockIdx.x * 256 + threadIdx.x;
  for (int e = threadIdx.x; e < LOUT * 2 * DSTATE; e += 256) {
    int l = e >> 5, c = e & 31;
    float v = xdbl[(size_t)(b * LOUT + l) * 256 + DTR + c];
    if (c < DSTATE) sB[l * DSTATE + c] = v;
    else            sC[l * DSTATE + (c - DSTATE)] = v;
  }
  __syncthreads();
  float a[DSTATE], s[DSTATE];
#pragma unroll
  for (int n = 0; n < DSTATE; ++n) {
    a[n] = -__expf(Alog[(size_t)d * DSTATE + n]);
    s[n] = 0.f;
  }
  float dpar = Dp[d];
  for (int l = 0; l < LOUT; ++l) {
    size_t r = (size_t)(b * LOUT + l);
    float dv = delta[r * DI + d];
    float xv = xbuf[r * DI + d];
    float dx = dv * xv;
    float y  = 0.f;
#pragma unroll
    for (int n = 0; n < DSTATE; ++n) {
      s[n] = __expf(dv * a[n]) * s[n] + dx * sB[l * DSTATE + n];
      y   += s[n] * sC[l * DSTATE + n];
    }
    y += xv * dpar;
    float zv = xz[r * XZ_N + DI + d];
    y *= zv / (1.f + __expf(-zv));               // y * silu(z)
    ybuf_h[r * DI + d] = (_Float16)y;
  }
}

// ---- classifier head: out[r,c] = hn[r,:] . fc_w[c,:] + fc_b[c] (fp32) ----
__global__ void __launch_bounds__(128)
logits_kernel(const float* __restrict__ hn, const float* __restrict__ fc_w,
              const float* __restrict__ fc_b, float* __restrict__ out) {
  int c = blockIdx.x, r = blockIdx.y;
  __shared__ float red[128];
  float acc = 0.f;
  for (int k = threadIdx.x; k < DM; k += 128)
    acc += hn[(size_t)r * DM + k] * fc_w[(size_t)c * DM + k];
  red[threadIdx.x] = acc;
  __syncthreads();
  for (int s = 64; s > 0; s >>= 1) {
    if (threadIdx.x < s) red[threadIdx.x] += red[threadIdx.x + s];
    __syncthreads();
  }
  if (threadIdx.x == 0) out[(size_t)r * NCLS + c] = red[0] + fc_b[c];
}

__global__ void zero_buf_h(_Float16* __restrict__ p, int n) {
  int i = blockIdx.x * 256 + threadIdx.x;
  if (i < n) p[i] = (_Float16)0.f;
}

// ---------------------------------------------------------------------------
extern "C" void kernel_launch(void* const* d_in, const int* in_sizes, int n_in,
                              void* d_out, int out_size, void* d_ws, size_t ws_size,
                              hipStream_t stream) {
  (void)in_sizes; (void)n_in; (void)out_size; (void)ws_size;
  const float* neuralInput = (const float*)d_in[0];
  const float* day_w       = (const float*)d_in[1];
  const float* day_b       = (const float*)d_in[2];
  const float* norm_w      = (const float*)d_in[3];
  const float* norm_b      = (const float*)d_in[4];
  const float* in_proj_w   = (const float*)d_in[5];
  const float* conv_w      = (const float*)d_in[6];
  const float* conv_b      = (const float*)d_in[7];
  const float* x_proj_w    = (const float*)d_in[8];
  const float* dt_proj_w   = (const float*)d_in[9];
  const float* dt_proj_b   = (const float*)d_in[10];
  const float* A_log       = (const float*)d_in[11];
  const float* D_param     = (const float*)d_in[12];
  const float* out_proj_w  = (const float*)d_in[13];
  const float* normf_w     = (const float*)d_in[14];
  const float* normf_b     = (const float*)d_in[15];
  const float* fc_w        = (const float*)d_in[16];
  const float* fc_b        = (const float*)d_in[17];
  const int*   dayIdx      = (const int*)d_in[18];

  // fp32 region
  float* ws = (float*)d_ws;
  size_t off = 0;
  float* tbuf  = ws + off; off += (size_t)BATCH * TLEN * ND;
  float* hbuf  = ws + off; off += (size_t)ROWS_PAD * DM;
  float* resid = ws + off; off += (size_t)ROWS_PAD * DM;
  float* hn    = ws + off; off += (size_t)ROWS_PAD * DM;
  float* xzbuf = ws + off; off += (size_t)ROWS_PAD * XZ_N;
  float* xbuf  = ws + off; off += (size_t)ROWS_PAD * DI;
  float* xdbl  = ws + off; off += (size_t)ROWS_PAD * 256;
  float* dbuf  = ws + off; off += (size_t)ROWS_PAD * DI;
  // f16 region (activation mirrors = GEMM A operands)
  _Float16* wsh = (_Float16*)(ws + off);
  size_t offh = 0;
  _Float16* hn_h   = wsh + offh; offh += (size_t)ROWS_PAD * DM;
  _Float16* xbuf_h = wsh + offh; offh += (size_t)ROWS_PAD * DI;
  _Float16* xdbl_h = wsh + offh; offh += (size_t)ROWS_PAD * 256;
  _Float16* ybuf_h = wsh + offh; offh += (size_t)ROWS_PAD * DI;   // total ~47 MB

  // Stage 1: day transform (softsign) + temporal unfold into padded row-major h.
  day_transform<<<dim3(TLEN, BATCH), ND, 0, stream>>>(neuralInput, day_w, day_b, dayIdx, tbuf);
  unfold_kernel<<<ROWS_PAD, 256, 0, stream>>>(tbuf, hbuf);
  {
    int n = (ROWS_PAD - ROWS) * DI;   // scan never writes pad rows of ybuf_h
    zero_buf_h<<<(n + 255) / 256, 256, 0, stream>>>(ybuf_h + (size_t)ROWS * DI, n);
  }

  for (int i = 0; i < 2; ++i) {
    ln_residual<<<ROWS_PAD, 256, 0, stream>>>(hbuf, resid, hn, hn_h,
        norm_w + (size_t)i * DM, norm_b + (size_t)i * DM, i == 0 ? 1 : 0);

    // xz = hn @ in_proj_w^T        (K=3584, N=14336) — dominant GEMM
    gemm_f16_wmma<<<dim3(XZ_N / 64, 2), 128, 0, stream>>>(
        hn_h, in_proj_w + (size_t)i * XZ_N * DM, xzbuf, nullptr, nullptr,
        DM, XZ_N, DM, DM, 0);

    conv_silu<<<ROWS_PAD, 256, 0, stream>>>(
        xzbuf, conv_w + (size_t)i * DI * DCONV, conv_b + (size_t)i * DI, xbuf, xbuf_h);

    // x_dbl = x @ x_proj_w^T       (K=7168, N=256); f16 mirror feeds dt GEMM
    gemm_f16_wmma<<<dim3(256 / 64, 2), 128, 0, stream>>>(
        xbuf_h, x_proj_w + (size_t)i * 256 * DI, xdbl, xdbl_h, nullptr,
        DI, 256, DI, DI, 0);

    // delta = softplus(x_dbl[:, :224] @ dt_proj_w^T + dt_proj_b)  (K=224, N=7168)
    gemm_f16_wmma<<<dim3(DI / 64, 2), 128, 0, stream>>>(
        xdbl_h, dt_proj_w + (size_t)i * DI * DTR, dbuf, nullptr,
        dt_proj_b + (size_t)i * DI, DTR, DI, 256, DTR, 1);

    scan_gate<<<dim3(DI / 256, BATCH), 256, 0, stream>>>(
        xdbl, dbuf, xbuf, xzbuf,
        A_log + (size_t)i * DI * DSTATE, D_param + (size_t)i * DI, ybuf_h);

    // h = y @ out_proj_w^T         (K=7168, N=3584)
    gemm_f16_wmma<<<dim3(DM / 64, 2), 128, 0, stream>>>(
        ybuf_h, out_proj_w + (size_t)i * DM * DI, hbuf, nullptr, nullptr,
        DI, DM, DI, DI, 0);
  }

  ln_residual<<<ROWS_PAD, 256, 0, stream>>>(hbuf, resid, hn, hn_h, normf_w, normf_b, 0);
  logits_kernel<<<dim3(NCLS, ROWS), 128, 0, stream>>>(hn, fc_w, fc_b, (float*)d_out);
}
